// LstmRnn_38139309588716
// MI455X (gfx1250) — compile-verified, matrix operand order checked by reading, not run on hardware
//
#include <hip/hip_runtime.h>

// ---------------------------------------------------------------------------
// LSTM warmup + autoregressive decode for MI455X (gfx1250, wave32).
// v_wmma_f32_16x16x32_bf16 matmuls; tiles staged into LDS by the Tensor Data
// Mover (TENSOR_LOAD_TO_LDS + s_wait_tensorcnt), with D#-generated LDS row
// padding for bank-conflict-free fragment reads. Falls back to per-lane
// global_load_async_to_lds_b128 if the TDM builtin is unavailable.
// ---------------------------------------------------------------------------

typedef __attribute__((ext_vector_type(8)))  __bf16 v8bf;
typedef __attribute__((ext_vector_type(16))) __bf16 v16bf;
typedef __attribute__((ext_vector_type(8)))  float  v8f;
typedef __attribute__((ext_vector_type(4)))  unsigned int v4u;
typedef __attribute__((ext_vector_type(8)))  int    v8i;
typedef __attribute__((ext_vector_type(4)))  int    v4i;

#define BSZ 4096
#define TT  72
#define FF  64
#define UU  256
#define U4  1024

// padded LDS K-strides (elements): conflict-free 16-lane x 16B fragment reads
#define LDK 264   // for K=256 rows (h, U): 128 DW/row + 4 DW pad
#define LDX 72    // for K=64 rows (x, W):  32 DW/row + 4 DW pad

// LDS byte offset = low 32 bits of the generic pointer (flat-LDS aperture).
__device__ __forceinline__ unsigned int lds_off(const void* p) {
  return (unsigned int)(unsigned long long)p;
}

#if __has_builtin(__builtin_amdgcn_tensor_load_to_lds)
#define USE_TDM 1
// Issue one TDM 2D tile load: rows x tile_k (bf16, K contiguous, row stride
// stride_elems), LDS destination padded by 4 DWORDs every 2^(pic+1) DWORDs.
// Wave-level instruction; descriptors are wave-uniform SGPR groups.
// clang-23 builtin arity: (v4u g0, v8i g1, v4i g2, v4i g3, v8i, i32 cpol).
__device__ __forceinline__ void tdm_load_2d(const __bf16* g, unsigned int ldst,
                                            int tile_k, int rows,
                                            long stride_elems, int pic) {
  const unsigned long long ga = (unsigned long long)g;
  v4u g0;
  g0[0] = 1u;                                   // count=1, user mode
  g0[1] = ldst;                                 // lds_addr (bytes)
  g0[2] = (unsigned int)ga;                     // global_addr[31:0]
  g0[3] = (unsigned int)((ga >> 32) & 0x01FFFFFFull) | (2u << 30);  // +type=2
  v8i g1;
  // data_size=2B(code1), pad_enable, pad_interval=pic, pad_amount=4DW(code3)
  g1[0] = (int)((1u << 16) | (1u << 20) | ((unsigned)pic << 22) | (3u << 25));
  g1[1] = (int)(((unsigned)tile_k & 0xFFFFu) << 16);            // tdim0 lo16
  g1[2] = (int)((((unsigned)rows & 0xFFFFu) << 16) |            // tdim1 lo16
                (((unsigned)tile_k >> 16) & 0xFFFFu));          // tdim0 hi16
  g1[3] = (int)((((unsigned)tile_k & 0xFFFFu) << 16) |          // tile_dim0
                (((unsigned)rows >> 16) & 0xFFFFu));            // tdim1 hi16
  g1[4] = (int)((unsigned)rows & 0xFFFFu);                      // tile_dim1
  g1[5] = (int)(unsigned int)(stride_elems & 0xFFFFFFFFl);      // stride lo32
  g1[6] = (int)(unsigned int)((stride_elems >> 32) & 0xFFFFl);  // stride hi16
  g1[7] = 0;
  const v4i gz4 = {0, 0, 0, 0};
  const v8i gz8 = {0, 0, 0, 0, 0, 0, 0, 0};
  __builtin_amdgcn_tensor_load_to_lds(g0, g1, gz4, gz4, gz8, 0);
}
__device__ __forceinline__ void tdm_wait_all() {
#if __has_builtin(__builtin_amdgcn_s_wait_tensorcnt)
  __builtin_amdgcn_s_wait_tensorcnt((short)0);
#else
  asm volatile("s_wait_tensorcnt 0x0" ::: "memory");
#endif
}
#else
#define USE_TDM 0
#endif

// Fallback: per-lane async copy of 16 bytes global -> LDS (ASYNCcnt).
__device__ __forceinline__ void async_ld_b128(const __bf16* lds_dst,
                                              const __bf16* gsrc) {
  unsigned int lo = lds_off(lds_dst);
  unsigned long long ga = (unsigned long long)(const void*)gsrc;
  asm volatile("global_load_async_to_lds_b128 %0, %1, off"
               :: "v"(lo), "v"(ga)
               : "memory");
}
__device__ __forceinline__ void async_wait_all() {
  asm volatile("s_wait_asynccnt 0x0" ::: "memory");
}

// Load a wave32 WMMA 16-bit fragment from a row-major (K-contiguous) row:
// lane = (half, lm); k elements = kbase..kbase+7 and kbase+16..kbase+23,
// kbase = half*8 within the 32-wide K chunk. Two 16B contiguous loads.
__device__ __forceinline__ v16bf load_frag(const __bf16* p) {
  v8bf lo = *(const v8bf*)(p);
  v8bf hi = *(const v8bf*)(p + 16);
  return __builtin_shufflevector(lo, hi, 0, 1, 2, 3, 4, 5, 6, 7,
                                         8, 9, 10, 11, 12, 13, 14, 15);
}

__device__ __forceinline__ v8f wmma_bf16(v16bf a, v16bf b, v8f c) {
  return __builtin_amdgcn_wmma_f32_16x16x32_bf16(
      /*neg_a=*/false, a, /*neg_b=*/false, b,
      /*c_mod=*/(short)0, c, /*reuse_a=*/false, /*reuse_b=*/false);
}

__device__ __forceinline__ float sigmoid_(float x) {
  return 1.0f / (1.0f + __expf(-x));
}
__device__ __forceinline__ float tanh_(float x) {
  return 1.0f - 2.0f / (__expf(2.0f * x) + 1.0f);
}

// ---------------------------------------------------------------------------
// One LSTM time step: z = X@W + Hin@U + b ; gates ; update C, Hout.
// grid (BSZ/64, UU/32), block 256 (8 waves = 4M x 2N). Each wave:
// 16 rows x 16 unit-cols x 4 gates. Tiles staged in LDS by TDM.
// Weights are N-major bf16: Wt[U4][FF], Ut[U4][UU] (K contiguous).
// ---------------------------------------------------------------------------
__global__ __launch_bounds__(256) void lstm_step_kernel(
    const __bf16* __restrict__ X, long ldx,
    const __bf16* __restrict__ Hin,
    const __bf16* __restrict__ Wt,    // [U4][FF]
    const __bf16* __restrict__ Ut,    // [U4][UU]
    const float*  __restrict__ bias,  // [U4]
    float*        __restrict__ C,     // [BSZ][UU] fp32 cell state (in place)
    __bf16*       __restrict__ Hout)  // [BSZ][UU]
{
  __shared__ __bf16 s_x[64 * LDX];        //  9.2 KB  x tile   [64][64]
  __shared__ __bf16 s_h[64 * LDK];        // 33.8 KB  h tile   [64][256]
  __shared__ __bf16 s_w[4 * 32 * LDX];    // 18.4 KB  W tiles  [4][32][64]
  __shared__ __bf16 s_u[4 * 32 * LDK];    // 67.6 KB  U tiles  [4][32][256]

  const int tid  = threadIdx.x;
  const int w    = tid >> 5;
  const int mblk = blockIdx.x * 64;
  const int nblk = blockIdx.y * 32;

#if USE_TDM
  // ---- stage tiles via Tensor Data Mover: 10 descriptors over 8 waves ----
  if (w < 4) {
    // U gate tile: 32 rows x 256 K, stride 256, pad 128+4 DW
    tdm_load_2d(Ut + (long)(w * UU + nblk) * UU, lds_off(&s_u[w * 32 * LDK]),
                UU, 32, UU, 6);
    if (w == 0)  // h tile: 64 rows x 256 K
      tdm_load_2d(Hin + (long)mblk * UU, lds_off(&s_h[0]), UU, 64, UU, 6);
    if (w == 1)  // x tile: 64 rows x 64 K, stride ldx, pad 32+4 DW
      tdm_load_2d(X + (long)mblk * ldx, lds_off(&s_x[0]), FF, 64, ldx, 4);
  } else {
    const int g = w - 4;  // W gate tile: 32 rows x 64 K
    tdm_load_2d(Wt + (long)(g * UU + nblk) * FF, lds_off(&s_w[g * 32 * LDX]),
                FF, 32, FF, 4);
  }
  tdm_wait_all();
#else
  // ---- fallback: per-lane async global->LDS (30 x b128 per lane) ----
  for (int q = tid; q < 64 * 32; q += 256) {
    const int row = q >> 5, k8 = (q & 31) << 3;
    async_ld_b128(&s_h[row * LDK + k8], Hin + (long)(mblk + row) * UU + k8);
  }
  for (int q = tid; q < 64 * 8; q += 256) {
    const int row = q >> 3, k8 = (q & 7) << 3;
    async_ld_b128(&s_x[row * LDX + k8], X + (long)(mblk + row) * ldx + k8);
  }
  for (int q = tid; q < 128 * 8; q += 256) {
    const int r = q >> 3, k8 = (q & 7) << 3;
    const int gn = (r >> 5) * UU + nblk + (r & 31);
    async_ld_b128(&s_w[r * LDX + k8], Wt + (long)gn * FF + k8);
  }
  for (int q = tid; q < 128 * 32; q += 256) {
    const int r = q >> 5, k8 = (q & 31) << 3;
    const int gn = (r >> 5) * UU + nblk + (r & 31);
    async_ld_b128(&s_u[r * LDK + k8], Ut + (long)gn * UU + k8);
  }
  async_wait_all();
#endif
  __syncthreads();

  // ---- WMMA main loop (fragments from LDS) ----
  const int lane = tid & 31;
  const int half = lane >> 4;
  const int lm   = lane & 15;
  const int koff = half * 8;
  const int mloc = (w & 3) * 16 + lm;   // A-fragment row (local)
  const int nloc = (w >> 2) * 16 + lm;  // B-fragment row (local, per gate)

  const v8f z = {0.f, 0.f, 0.f, 0.f, 0.f, 0.f, 0.f, 0.f};
  v8f acc0 = z, acc1 = z, acc2 = z, acc3 = z;  // i, f, g, o

  const __bf16* xr = &s_x[mloc * LDX];
  const __bf16* hr = &s_h[mloc * LDK];
  const __bf16* w0 = &s_w[(0 * 32 + nloc) * LDX];
  const __bf16* w1 = &s_w[(1 * 32 + nloc) * LDX];
  const __bf16* w2 = &s_w[(2 * 32 + nloc) * LDX];
  const __bf16* w3 = &s_w[(3 * 32 + nloc) * LDX];
  const __bf16* u0 = &s_u[(0 * 32 + nloc) * LDK];
  const __bf16* u1 = &s_u[(1 * 32 + nloc) * LDK];
  const __bf16* u2 = &s_u[(2 * 32 + nloc) * LDK];
  const __bf16* u3 = &s_u[(3 * 32 + nloc) * LDK];

#pragma unroll
  for (int kc = 0; kc < FF; kc += 32) {
    const int o = kc + koff;
    // hoist all 10 ds_loads of this chunk ahead of the WMMA group
    const v16bf a  = load_frag(xr + o);
    const v16bf b0 = load_frag(w0 + o);
    const v16bf b1 = load_frag(w1 + o);
    const v16bf b2 = load_frag(w2 + o);
    const v16bf b3 = load_frag(w3 + o);
    acc0 = wmma_bf16(a, b0, acc0);
    acc1 = wmma_bf16(a, b1, acc1);
    acc2 = wmma_bf16(a, b2, acc2);
    acc3 = wmma_bf16(a, b3, acc3);
  }
#pragma unroll
  for (int kc = 0; kc < UU; kc += 32) {
    const int o = kc + koff;
    const v16bf a  = load_frag(hr + o);
    const v16bf b0 = load_frag(u0 + o);
    const v16bf b1 = load_frag(u1 + o);
    const v16bf b2 = load_frag(u2 + o);
    const v16bf b3 = load_frag(u3 + o);
    acc0 = wmma_bf16(a, b0, acc0);
    acc1 = wmma_bf16(a, b1, acc1);
    acc2 = wmma_bf16(a, b2, acc2);
    acc3 = wmma_bf16(a, b3, acc3);
  }

  // ---- gate epilogue: C/D layout lane lm = col, VGPR r = row r + 8*half ----
  const int col = nblk + (w >> 2) * 16 + lm;
  const float bi  = bias[col];
  const float bf_ = bias[col + UU];
  const float bg  = bias[col + 2 * UU];
  const float bo  = bias[col + 3 * UU];
#pragma unroll
  for (int r = 0; r < 8; ++r) {
    const int  row = mblk + (w & 3) * 16 + half * 8 + r;
    const long idx = (long)row * UU + col;
    const float iv = sigmoid_(acc0[r] + bi);
    const float fv = sigmoid_(acc1[r] + bf_);
    const float gv = tanh_(acc2[r] + bg);
    const float ov = sigmoid_(acc3[r] + bo);
    const float cn = fv * C[idx] + iv * gv;
    C[idx]    = cn;
    Hout[idx] = (__bf16)(ov * tanh_(cn));
  }
}

// ---------------------------------------------------------------------------
// pred = H @ Wd + bd. grid (BSZ/32), block 256 (8 waves = 2M x 4N).
// Small K=256 GEMM, direct global loads (weights L2/L0 resident).
// ---------------------------------------------------------------------------
__global__ __launch_bounds__(256) void dense_kernel(
    const __bf16* __restrict__ H,    // [BSZ][UU]
    const __bf16* __restrict__ Wdt,  // [FF][UU] N-major
    const float*  __restrict__ bd,   // [FF]
    float*        __restrict__ out,  // [BSZ][out_steps][FF]
    int tstep, int out_steps,
    __bf16*       __restrict__ Pred) // [BSZ][FF]
{
  const int tid   = threadIdx.x;
  const int w     = tid >> 5;
  const int lane  = tid & 31;
  const int half  = lane >> 4;
  const int lm    = lane & 15;
  const int mrow0 = blockIdx.x * 32 + (w & 1) * 16;
  const int ncol0 = (w >> 1) * 16;
  const int koff  = half * 8;

  v8f acc = {0.f, 0.f, 0.f, 0.f, 0.f, 0.f, 0.f, 0.f};
  const __bf16* hrow = H + (long)(mrow0 + lm) * UU;
  const __bf16* wrow = Wdt + (long)(ncol0 + lm) * UU;
#pragma unroll
  for (int kc = 0; kc < UU; kc += 32) {
    const v16bf a = load_frag(hrow + kc + koff);
    const v16bf b = load_frag(wrow + kc + koff);
    acc = wmma_bf16(a, b, acc);
  }

  const int col = ncol0 + lm;
  const float bb = bd[col];
#pragma unroll
  for (int r = 0; r < 8; ++r) {
    const int row = mrow0 + half * 8 + r;
    const float v = acc[r] + bb;
    out[(long)row * out_steps * FF + (long)tstep * FF + col] = v;
    Pred[(long)row * FF + col] = (__bf16)v;
  }
}

// ---------------------------------------------------------------------------
// Prep kernels (run every launch; deterministic).
// ---------------------------------------------------------------------------
__global__ void transpose_to_bf16(const float* __restrict__ src,  // [R][Cc]
                                  __bf16* __restrict__ dst,       // [Cc][R]
                                  int R, int Cc) {
  long i = (long)blockIdx.x * blockDim.x + threadIdx.x;
  if (i < (long)R * Cc) {
    int r = (int)(i / Cc);
    int c = (int)(i % Cc);
    dst[(long)c * R + r] = (__bf16)src[i];
  }
}

__global__ void convert_bf16(const float* __restrict__ src,
                             __bf16* __restrict__ dst, long n) {
  long i = (long)blockIdx.x * blockDim.x + threadIdx.x;
  if (i < n) dst[i] = (__bf16)src[i];
}

__global__ void zero_state(float* __restrict__ C, __bf16* __restrict__ H,
                           long n) {
  long i = (long)blockIdx.x * blockDim.x + threadIdx.x;
  if (i < n) {
    C[i] = 0.f;
    H[i] = (__bf16)0.f;
  }
}

// ---------------------------------------------------------------------------
extern "C" void kernel_launch(void* const* d_in, const int* in_sizes, int n_in,
                              void* d_out, int out_size, void* d_ws,
                              size_t ws_size, hipStream_t stream) {
  const float* inputs = (const float*)d_in[0];  // [BSZ][TT][FF]
  const float* W      = (const float*)d_in[1];  // [FF][U4]
  const float* U      = (const float*)d_in[2];  // [UU][U4]
  const float* b      = (const float*)d_in[3];  // [U4]
  const float* Wd     = (const float*)d_in[4];  // [UU][FF]
  const float* bd     = (const float*)d_in[5];  // [FF]
  float* out = (float*)d_out;
  const int out_steps = out_size / (BSZ * FF);  // 24

  // Workspace carve-up (all chunk sizes are multiples of 256 bytes).
  char* p = (char*)d_ws;
  __bf16* Wt   = (__bf16*)p; p += (long)U4 * FF * 2;        // 128 KB
  __bf16* Ut   = (__bf16*)p; p += (long)U4 * UU * 2;        // 512 KB
  __bf16* Wdt  = (__bf16*)p; p += (long)FF * UU * 2;        //  32 KB
  __bf16* Xbf  = (__bf16*)p; p += (long)BSZ * TT * FF * 2;  //  36 MB
  __bf16* H0   = (__bf16*)p; p += (long)BSZ * UU * 2;       //   2 MB
  __bf16* H1   = (__bf16*)p; p += (long)BSZ * UU * 2;       //   2 MB
  float*  C    = (float*)p;  p += (long)BSZ * UU * 4;       //   4 MB
  __bf16* Pred = (__bf16*)p; p += (long)BSZ * FF * 2;       // 512 KB

  // ---- prep: weight transposes to N-major bf16, input cast, state zero ----
  {
    long n = (long)FF * U4;
    transpose_to_bf16<<<(unsigned)((n + 255) / 256), 256, 0, stream>>>(W, Wt, FF, U4);
  }
  {
    long n = (long)UU * U4;
    transpose_to_bf16<<<(unsigned)((n + 255) / 256), 256, 0, stream>>>(U, Ut, UU, U4);
  }
  {
    long n = (long)UU * FF;
    transpose_to_bf16<<<(unsigned)((n + 255) / 256), 256, 0, stream>>>(Wd, Wdt, UU, FF);
  }
  {
    long n = (long)BSZ * TT * FF;
    convert_bf16<<<(unsigned)((n + 255) / 256), 256, 0, stream>>>(inputs, Xbf, n);
  }
  {
    long n = (long)BSZ * UU;
    zero_state<<<(unsigned)((n + 255) / 256), 256, 0, stream>>>(C, H0, n);
  }

  const dim3 sgrid(BSZ / 64, UU / 32);
  __bf16* hin  = H0;
  __bf16* hout = H1;

  // ---- warmup over the input sequence ----
  for (int t = 0; t < TT; ++t) {
    lstm_step_kernel<<<sgrid, 256, 0, stream>>>(
        Xbf + (long)t * FF, (long)TT * FF, hin, Wt, Ut, b, C, hout);
    __bf16* tmp = hin; hin = hout; hout = tmp;
  }

  // ---- pred0 ----
  dense_kernel<<<BSZ / 32, 256, 0, stream>>>(hin, Wdt, bd, out, 0, out_steps, Pred);

  // ---- autoregressive decode ----
  for (int s = 1; s < out_steps; ++s) {
    lstm_step_kernel<<<sgrid, 256, 0, stream>>>(
        Pred, (long)FF, hin, Wt, Ut, b, C, hout);
    __bf16* tmp = hin; hin = hout; hout = tmp;
    dense_kernel<<<BSZ / 32, 256, 0, stream>>>(hin, Wdt, bd, out, s, out_steps, Pred);
  }
}
